// SimpleGATLayer_28630251995683
// MI455X (gfx1250) — matrix operand order repaired and checked by reading, not compiled, not run on hardware
//
#include <hip/hip_runtime.h>

// ---- problem constants (from reference) ----
#define NN      10          // nodes per graph
#define DIN     256
#define DOUT    256
#define NH      4           // heads
#define HD      64          // head dim
#define ALPHA   (-0.2f)
#define EPS_LOG 1e-6f
#define EPS_LN  1e-5f

// ---- tiling ----
#define BPW       4                   // batches per workgroup
#define ROWS      (BPW * NN)          // 40 real GEMM rows
#define ROWS_PAD  48                  // 3 M-tiles of 16 (pad rows are don't-care)
#define MT        3                   // M tiles
#define KCH       32                  // K per WMMA (bf16)
#define NKC       (DIN / KCH)         // 8 K chunks
#define CHUNK_F32 (ROWS_PAD * KCH)    // 1536 f32 = 6144 B per staged chunk
#define NBUF      3                   // triple-buffered staging

typedef __attribute__((ext_vector_type(16))) __bf16 v16bf;
typedef __attribute__((ext_vector_type(8)))  float  v8f;

// ------------------------------------------------------------------
// Kernel 0: convert W_w (row-major [DOUT][DIN], f32) to bf16 workspace.
// Row-major bf16 => per-lane contiguous-K B fragments, L2-resident (128 KB).
// ------------------------------------------------------------------
__global__ void gat_wconv(const float* __restrict__ W, __bf16* __restrict__ Wt, int n) {
    int i = blockIdx.x * blockDim.x + threadIdx.x;
    if (i < n) Wt[i] = (__bf16)W[i];
}

// ------------------------------------------------------------------
// Async-copy one 48x32 f32 x-chunk into an LDS staging buffer.
// Exactly 3 GLOBAL_LOAD_ASYNC_TO_LDS_B64 per thread => ASYNCcnt +3 per wave.
// ------------------------------------------------------------------
__device__ __forceinline__ void prefetch_chunk(const float* __restrict__ x,
                                               int b0, int kc,
                                               float* dst, int tid, long maxrow) {
#pragma unroll
    for (int u = 0; u < 3; ++u) {
        int t   = tid + u * 256;           // 0..767
        int row = t >> 4;                  // 0..47 (rows 40..47 = pad, clamped)
        int sub = t & 15;                  // 16 x 8-byte pieces per row
        long grow = (long)b0 * NN + row;
        if (grow > maxrow) grow = maxrow;  // keep pad reads in-bounds
        const float* g = x + grow * DIN + (long)kc * KCH + sub * 2;
        unsigned lds   = (unsigned)(unsigned long long)(const void*)(dst + t * 2);
        asm volatile("global_load_async_to_lds_b64 %0, %1, off"
                     :: "v"(lds), "v"((unsigned long long)(size_t)g)
                     : "memory");
    }
}

// ------------------------------------------------------------------
// Kernel 1: fused projection GEMM (bf16 WMMA, f32 accum, async-LDS
// triple-buffered x stream) + GAT row-0 attention + softmax + LN.
// ------------------------------------------------------------------
__global__ __launch_bounds__(256)
void gat_fused(const float* __restrict__ x,
               const float* __restrict__ adj_phys,
               const float* __restrict__ Wb,
               const float* __restrict__ a,
               const float* __restrict__ ln_g,
               const float* __restrict__ ln_b,
               const __bf16* __restrict__ Wt,
               float* __restrict__ out,
               int Btot) {
    __shared__ float  Axf[NBUF * CHUNK_F32];     // 18 KB triple-buffered x (f32)
    __shared__ float  hbuf[ROWS * DOUT];         // 40 KB h = xW^T + b
    __shared__ float  eiL[BPW * NH];
    __shared__ float  ejL[BPW * NH * NN];
    __shared__ float  attnL[BPW * NH * NN];
    __shared__ float  redS[256], redQ[256];
    __shared__ float  muL[BPW], rinvL[BPW];

    const int tid  = threadIdx.x;
    const int wave = tid >> 5;
    const int lane = tid & 31;
    const int b0   = blockIdx.x * BPW;
    if (b0 >= Btot) return;
    const long maxrow = (long)Btot * NN - 1;

    // f32 accumulators: 3 M-tiles x 2 N-tiles per wave
    v8f acc[MT][2];
    v8f vzero = {0.f, 0.f, 0.f, 0.f, 0.f, 0.f, 0.f, 0.f};
#pragma unroll
    for (int mt = 0; mt < MT; ++mt) { acc[mt][0] = vzero; acc[mt][1] = vzero; }

    const int nbase = wave * 2;                  // this wave's first N tile

    // software pipeline depth 2: chunks 0 and 1 in flight before the loop
    prefetch_chunk(x, b0, 0, Axf,             tid, maxrow);
    prefetch_chunk(x, b0, 1, Axf + CHUNK_F32, tid, maxrow);

    int cur = 0;                                 // rotating buffer index
    for (int kc = 0; kc < NKC; ++kc) {
        // own chunk-kc async ops retired (in-order; newest 3 = chunk kc+1)
        if (kc == NKC - 1) asm volatile("s_wait_asynccnt 0x0" ::: "memory");
        else               asm volatile("s_wait_asynccnt 0x3" ::: "memory");
        __syncthreads();                         // everyone's chunk kc landed

        // prefetch distance 2: all readers of buf[(kc+2)%3] finished before
        // the barrier above, so it is safe to overwrite now.
        if (kc + 2 < NKC) {
            int nb2 = cur + 2; if (nb2 >= NBUF) nb2 -= NBUF;
            prefetch_chunk(x, b0, kc + 2, Axf + nb2 * CHUNK_F32, tid, maxrow);
        }

        const float* __restrict__ cbuf = Axf + cur * CHUNK_F32;

        // ---- B fragments straight from L2-resident bf16 W ----
        // lane: column n = lane%16, contiguous K group 16*(lane/16)
        v16bf bfrag[2];
#pragma unroll
        for (int nb = 0; nb < 2; ++nb) {
            int n  = (nbase + nb) * 16 + (lane & 15);
            int k0 = kc * KCH + ((lane >> 4) << 4);
            const __bf16* p = Wt + (long)n * DIN + k0;
#pragma unroll
            for (int i = 0; i < 16; ++i) bfrag[nb][i] = p[i];
        }

        // ---- build ALL 3 A fragments (f32 from LDS -> packed bf16 cvts) ----
        v16bf afr[MT];
#pragma unroll
        for (int mt = 0; mt < MT; ++mt) {
            int row  = mt * 16 + (lane & 15);
            int base = (lane >> 4) << 4;
            const float4 f0 = *(const float4*)(cbuf + row * KCH + base);
            const float4 f1 = *(const float4*)(cbuf + row * KCH + base + 4);
            const float4 f2 = *(const float4*)(cbuf + row * KCH + base + 8);
            const float4 f3 = *(const float4*)(cbuf + row * KCH + base + 12);
            afr[mt][0]  = (__bf16)f0.x; afr[mt][1]  = (__bf16)f0.y;
            afr[mt][2]  = (__bf16)f0.z; afr[mt][3]  = (__bf16)f0.w;
            afr[mt][4]  = (__bf16)f1.x; afr[mt][5]  = (__bf16)f1.y;
            afr[mt][6]  = (__bf16)f1.z; afr[mt][7]  = (__bf16)f1.w;
            afr[mt][8]  = (__bf16)f2.x; afr[mt][9]  = (__bf16)f2.y;
            afr[mt][10] = (__bf16)f2.z; afr[mt][11] = (__bf16)f2.w;
            afr[mt][12] = (__bf16)f3.x; afr[mt][13] = (__bf16)f3.y;
            afr[mt][14] = (__bf16)f3.z; afr[mt][15] = (__bf16)f3.w;
        }

        // Pin all fragments live here: forces distinct register octets,
        // clusters the ds_loads/cvts ahead of the WMMA burst, and keeps the
        // scheduler from sinking fragment builds between WMMAs.
        asm volatile("" : "+v"(afr[0]), "+v"(afr[1]), "+v"(afr[2]),
                          "+v"(bfrag[0]), "+v"(bfrag[1]));

        // ---- 6 back-to-back WMMAs ----
#pragma unroll
        for (int mt = 0; mt < MT; ++mt) {
            acc[mt][0] = __builtin_amdgcn_wmma_f32_16x16x32_bf16(
                false, afr[mt], false, bfrag[0], (short)0, acc[mt][0], false, false);
            acc[mt][1] = __builtin_amdgcn_wmma_f32_16x16x32_bf16(
                false, afr[mt], false, bfrag[1], (short)0, acc[mt][1], false, false);
        }

        cur = cur + 1; if (cur >= NBUF) cur -= NBUF;
    }
    __syncthreads();

    // ---- write h (+bias) to LDS; C/D layout: VGPR v -> M = v + 8*(lane/16) ----
#pragma unroll
    for (int nb = 0; nb < 2; ++nb) {
        int col   = (nbase + nb) * 16 + (lane & 15);
        float bia = Wb[col];
#pragma unroll
        for (int v = 0; v < 8; ++v) {
            int row = ((lane >> 4) << 3) + v;    // within-tile row
#pragma unroll
            for (int mt = 0; mt < MT; ++mt) {
                int r = mt * 16 + row;
                if (r < ROWS) hbuf[r * DOUT + col] = acc[mt][nb][v] + bia;
            }
        }
    }
    __syncthreads();

    // ================= phase 2: attention row 0 + LN =================
    // ei/ej dots: 160 threads, one (b,h,j) each
    if (tid < BPW * NH * NN) {
        int bl  = tid / (NH * NN);
        int rem = tid % (NH * NN);
        int h   = rem / NN;
        int j   = rem % NN;
        const float* av = a + h * 2 * HD;        // a[h][0..127]
        const float* hr = hbuf + (bl * NN + j) * DOUT + h * HD;
        float ej = 0.f;
        for (int d = 0; d < HD; ++d) ej += hr[d] * av[HD + d];
        ejL[tid] = ej;
        if (j == 0) {
            float ei = 0.f;
            for (int d = 0; d < HD; ++d) ei += hr[d] * av[d];
            eiL[bl * NH + h] = ei;
        }
    }
    __syncthreads();

    // leaky + log(adj row 0) + softmax over 10 nodes: 16 threads
    if (tid < BPW * NH) {
        int bl = tid / NH;
        float ei = eiL[tid];
        float e[NN], m = -1e30f;
#pragma unroll
        for (int j = 0; j < NN; ++j) {
            float v = ei + ejL[tid * NN + j];
            v = (v >= 0.f) ? v : ALPHA * v;
            float adjv = (j == 0) ? 1.0f
                                  : adj_phys[(long)(b0 + bl) * (NN - 1) + (j - 1)];
            v += logf(adjv + EPS_LOG);
            e[j] = v;
            m = fmaxf(m, v);
        }
        float s = 0.f;
#pragma unroll
        for (int j = 0; j < NN; ++j) { e[j] = expf(e[j] - m); s += e[j]; }
        float inv = 1.f / s;
#pragma unroll
        for (int j = 0; j < NN; ++j) attnL[tid * NN + j] = e[j] * inv;
    }
    __syncthreads();

    // weighted sum over nodes + LayerNorm; thread owns 4 features of 1 batch
    const int bl = tid >> 6;
    const int f0 = (tid & 63) * 4;
    const int hh = f0 >> 6;                      // 4 consecutive feats share a head
    const float* attn = attnL + (bl * NH + hh) * NN;
    float o[4], s = 0.f, ss = 0.f;
#pragma unroll
    for (int c = 0; c < 4; ++c) {
        int f = f0 + c;
        float acc2 = 0.f;
#pragma unroll
        for (int j = 0; j < NN; ++j) acc2 += attn[j] * hbuf[(bl * NN + j) * DOUT + f];
        o[c] = acc2; s += acc2; ss += acc2 * acc2;
    }
    redS[tid] = s; redQ[tid] = ss;
    __syncthreads();
    if (tid < BPW) {
        float s2 = 0.f, q2 = 0.f;
        for (int i = 0; i < 64; ++i) { s2 += redS[tid * 64 + i]; q2 += redQ[tid * 64 + i]; }
        float mu  = s2 * (1.0f / DOUT);
        float var = q2 * (1.0f / DOUT) - mu * mu;
        muL[tid]   = mu;
        rinvL[tid] = rsqrtf(var + EPS_LN);
    }
    __syncthreads();
    const float mu = muL[bl], ri = rinvL[bl];
#pragma unroll
    for (int c = 0; c < 4; ++c) {
        int f = f0 + c;
        out[(long)(b0 + bl) * DOUT + f] = ln_g[f] * (o[c] - mu) * ri + ln_b[f];
    }
}

// ------------------------------------------------------------------
extern "C" void kernel_launch(void* const* d_in, const int* in_sizes, int n_in,
                              void* d_out, int out_size, void* d_ws, size_t ws_size,
                              hipStream_t stream) {
    const float* x        = (const float*)d_in[0];
    const float* adj_phys = (const float*)d_in[1];
    const float* W_w      = (const float*)d_in[2];
    const float* W_b      = (const float*)d_in[3];
    const float* a        = (const float*)d_in[4];
    const float* ln_g     = (const float*)d_in[5];
    const float* ln_b     = (const float*)d_in[6];
    float* out            = (float*)d_out;

    const int Btot = in_sizes[0] / (NN * DIN);

    __bf16* Wt = (__bf16*)d_ws;                  // 128 KB bf16 weights
    const int wn = DOUT * DIN;
    gat_wconv<<<(wn + 255) / 256, 256, 0, stream>>>(W_w, Wt, wn);

    const int nblocks = (Btot + BPW - 1) / BPW;  // 8192 for B=32768
    gat_fused<<<nblocks, 256, 0, stream>>>(x, adj_phys, W_b, a, ln_g, ln_b,
                                           Wt, out, Btot);
}
